// Interventions_5935644803128
// MI455X (gfx1250) — matrix operand order repaired and checked by reading, not compiled, not run on hardware
//
#include <hip/hip_runtime.h>
#include <hip/hip_bf16.h>
#include <stdint.h>

// Problem constants (match reference)
#define B_TOT 8192
#define Dd    64
#define Kk    16
#define Hh    5
#define INN   256     // IN = D*P
#define NN    192     // packed output cols per group j: 80 + 16 + 80 + 16
#define MT    64      // rows of B handled per block

// LDS strides (padded to rotate banks, keep 16B alignment)
#define SA_STRIDE 264     // ushorts per X row (256 + 8)
#define SB_STRIDE 40      // ushorts per W row (32 + 8 pad = 20 dwords)
#define SS_STRIDE 196     // floats per score row (192 + 4)
#define SB_OFF    33792   // sA = [0, 33792)
#define SB_BUF_BYTES (NN * SB_STRIDE * 2)     // 15360 per slice buffer
#define SMEM_BYTES (SB_OFF + 2 * SB_BUF_BYTES) // 64512 total; scores alias [0,50176)

typedef float          v8f     __attribute__((ext_vector_type(8)));
typedef __bf16         bf16x16 __attribute__((ext_vector_type(16)));
typedef unsigned short v8u     __attribute__((ext_vector_type(8)));
typedef unsigned int   u32x4   __attribute__((ext_vector_type(4)));
typedef int            i32x4   __attribute__((ext_vector_type(4)));
typedef int            i32x8   __attribute__((ext_vector_type(8)));

#if defined(__has_builtin)
#  if __has_builtin(__builtin_amdgcn_tensor_load_to_lds)
#    define HAVE_TDM 1
#  endif
#endif
#if defined(__has_include)
#  if __has_include(<hip/amd_detail/amd_gfx1250_TDM.h>)
#    define TDM_SIX_ARGS 1   // therock-10.0 headers => clang-23 6-arg builtin
#  endif
#endif

__device__ __forceinline__ unsigned short f2bf(float f) {
  unsigned int u = __float_as_uint(f);
  u += 0x7FFFu + ((u >> 16) & 1u);           // round-to-nearest-even
  return (unsigned short)(u >> 16);
}

#ifdef HAVE_TDM
// TDM: DMA one 192x32 bf16 weight k-slice (tile of the 192x256 tensor Wc[j])
// into LDS with a 4-dword pad after each 16-dword row (SB_STRIDE layout).
// Descriptor per CDNA5 ISA ch.8 (D# groups 0/1; 2D tensor so groups 2/3 zero).
__device__ __forceinline__ void tdm_load_b_slice(const unsigned short* Wc,
                                                 int j, int kk,
                                                 uint32_t lds_byte_addr) {
  const uint64_t ga =
      (uint64_t)(uintptr_t)(Wc + (size_t)j * NN * INN + (size_t)kk * 32);
  u32x4 g0;
  g0[0] = 1u;                                           // count=1, user mode
  g0[1] = lds_byte_addr;                                // lds_addr [63:32]
  g0[2] = (uint32_t)ga;                                 // global_addr[31:0]
  g0[3] = (uint32_t)((ga >> 32) & 0x01FFFFFFu) | (2u << 30);  // [56:32] | type=2
  i32x8 g1;
  g1[0] = (1 << 16)      // data_size = 2 bytes
        | (1 << 20)      // pad_enable
        | (3 << 22)      // pad_interval: 16 dwords (one 32-elem bf16 row)
        | (3 << 25);     // pad_amount: 4 dwords (8 bf16 of pad)
  g1[1] = (int)((INN & 0xFFFF) << 16);   // tensor_dim0[15:0] @ bits 63:48
  g1[2] = (int)(NN << 16);               // tensor_dim1[15:0] @ bits 95:80
  g1[3] = (int)(32 << 16);               // tile_dim0 = 32    @ bits 127:112
  g1[4] = NN;                            // tile_dim1 = 192   @ bits 143:128
  g1[5] = INN;                           // tensor_dim0_stride[31:0]
  g1[6] = 0;
  g1[7] = 0;
  i32x4 z4; z4[0] = 0; z4[1] = 0; z4[2] = 0; z4[3] = 0;
#ifdef TDM_SIX_ARGS
  i32x8 z8; for (int t = 0; t < 8; ++t) z8[t] = 0;
  __builtin_amdgcn_tensor_load_to_lds(g0, g1, z4, z4, z8, 0);
#else
  __builtin_amdgcn_tensor_load_to_lds(g0, g1, z4, z4, 0);
#endif
}
#endif

// ---------------------------------------------------------------------------
// Pack W1/Wx (mean) and sW1/sWx (shared) into bf16 Wc[j][n][i], n in [0,192)
// ---------------------------------------------------------------------------
__global__ __launch_bounds__(256) void prep_weights(
    const float* __restrict__ W1, const float* __restrict__ Wx,
    const float* __restrict__ sW1, const float* __restrict__ sWx,
    unsigned short* __restrict__ Wc)
{
  const int j = blockIdx.x;
  for (int e = threadIdx.x; e < NN * INN; e += 256) {
    const int n = e >> 8, i = e & 255;
    float v;
    if (n < 80)        v = W1 [(((n      / 5) * Dd + j) * Hh + (n      % 5)) * INN + i];
    else if (n < 96)   v = Wx [(( n - 80)      * Dd + j)                     * INN + i];
    else if (n < 176)  v = sW1[((((n - 96) / 5) * Dd + j) * Hh + ((n - 96) % 5)) * INN + i];
    else               v = sWx[(( n - 176)     * Dd + j)                     * INN + i];
    Wc[((size_t)j * NN + n) * INN + i] = f2bf(v);
  }
}

// ---------------------------------------------------------------------------
// Row-masked softmax weights w[j][i][k] from targets (D,K,K-1)
// ---------------------------------------------------------------------------
__global__ __launch_bounds__(256) void prep_softmax(
    const float* __restrict__ targets, float* __restrict__ wsoft)
{
  const int p = blockIdx.x * 256 + threadIdx.x;   // 0..1023 = (j,i)
  if (p >= Dd * Kk) return;
  const int j = p >> 4, irow = p & 15;
  float base[Kk];
  float m = -1e30f;
  for (int c = 0; c <= irow; ++c) {
    float b = (c == irow) ? 1.0f : 0.0f;
    if (c >= 1) b += targets[(j * Kk + irow) * (Kk - 1) + (c - 1)];
    base[c] = b;
    m = fmaxf(m, b);
  }
  float s = 0.0f;
  for (int c = 0; c <= irow; ++c) { base[c] = __expf(base[c] - m); s += base[c]; }
  const float inv = 1.0f / s;
  for (int c = 0; c < Kk; ++c)
    wsoft[p * Kk + c] = (c <= irow) ? base[c] * inv : 0.0f;
}

// ---------------------------------------------------------------------------
// Fused grouped-MLP (mean+shared) + Gaussian NLL + masked-softmax mix.
// WMMA bf16 16x16x32 GEMM core; B-slices staged via TDM (double-buffered).
// ---------------------------------------------------------------------------
__global__ __launch_bounds__(256) void fused_mlp_nll(
    const float* __restrict__ xv,   const float* __restrict__ xin,
    const float* __restrict__ b1,   const float* __restrict__ Wh,
    const float* __restrict__ b2,   const float* __restrict__ sb1,
    const float* __restrict__ sWh,  const float* __restrict__ sb2,
    const unsigned short* __restrict__ Wc, const float* __restrict__ wsoft,
    float* __restrict__ out)
{
  __shared__ __align__(16) unsigned char smem[SMEM_BYTES];
  unsigned short* sA      = (unsigned short*)smem;              // [64][264] bf16
  unsigned short* sB      = (unsigned short*)(smem + SB_OFF);   // 2x[192][40] bf16
  float*          sScores = (float*)smem;                       // [64][196] (alias)
  uint32_t*       sA32    = (uint32_t*)sA;
#ifndef HAVE_TDM
  uint32_t*       sB32    = (uint32_t*)sB;
#endif

  const int tid    = threadIdx.x;
  const int lane   = tid & 31;
  const int wave   = tid >> 5;          // 8 wave32s
  const int m_tile = wave & 3;          // 4 M sub-tiles of 16 rows
  const int n_half = wave >> 2;         // 2 halves of 12 N-tiles
  const int l16    = lane & 15;
  const int lhalf  = lane >> 4;
  const int b0     = blockIdx.x * MT;

  const int erow = tid & 63;            // epilogue: row owned by this thread
  const int eq   = tid >> 6;            // epilogue: k-quarter (k = eq + 4*kq)

#ifdef HAVE_TDM
  // low 32 bits of the flat shared pointer == LDS byte address (aperture
  // upper bits live in [63:32]); used for the TDM descriptor lds_addr.
  const uint32_t sB_lds = (uint32_t)(uintptr_t)sB;
#endif

  float acc[Kk];
  #pragma unroll
  for (int i = 0; i < Kk; ++i) acc[i] = 0.0f;

  for (int j = 0; j < Dd; ++j) {
    // --- stage X tile (fp32 -> bf16) + prefetch next group's tile ---
    #pragma unroll
    for (int t = 0; t < 16; ++t) {
      const int flat = tid + (t << 8);
      const int r = flat >> 6, c4 = flat & 63;
      const size_t src = ((size_t)(b0 + r) * Dd + j) * INN + (c4 << 2);
      const float4 f4 = *(const float4*)(xin + src);
      const uint32_t lo0 = ((uint32_t)f2bf(f4.y) << 16) | f2bf(f4.x);
      const uint32_t lo1 = ((uint32_t)f2bf(f4.w) << 16) | f2bf(f4.z);
      const int di = r * (SA_STRIDE / 2) + (c4 << 1);
      sA32[di]     = lo0;
      sA32[di + 1] = lo1;
      if (j + 1 < Dd)
        __builtin_prefetch(xin + src + INN, 0, 0);   // global_prefetch_b8
    }

#ifdef HAVE_TDM
    if (wave == 0)                                    // one DMA issue per block
      tdm_load_b_slice(Wc, j, 0, sB_lds);
#endif
    __syncthreads();

    // --- GEMM: C(64x192) += Xj(64x256) * Wj^T(256x192), bf16 WMMA ---
    v8f cacc[6];
    #pragma unroll
    for (int t = 0; t < 6; ++t)
      cacc[t] = (v8f){0.f, 0.f, 0.f, 0.f, 0.f, 0.f, 0.f, 0.f};

    #pragma unroll
    for (int kk = 0; kk < 8; ++kk) {            // K = 8 x 32
      unsigned short* sBk = sB + (kk & 1) * (NN * SB_STRIDE);
#ifdef HAVE_TDM
      // issue next slice DMA, then wait: current slice landed (<=1 in flight)
      if (wave == 0 && kk + 1 < 8)
        tdm_load_b_slice(Wc, j, kk + 1, sB_lds + ((kk + 1) & 1) * SB_BUF_BYTES);
      if (kk + 1 < 8) __builtin_amdgcn_s_wait_tensorcnt(1);
      else            __builtin_amdgcn_s_wait_tensorcnt(0);
#else
      // fallback: synchronous cooperative copy of the 192x32 slice
      #pragma unroll
      for (int t = 0; t < 6; ++t) {
        const int flat = tid + (t << 8);        // 1536 uint2
        const int n = flat >> 3, cq = flat & 7;
        const uint2 v = *(const uint2*)((const uint32_t*)Wc +
                         (((size_t)j * NN + n) << 7) + (kk << 4) + (cq << 1));
        *(uint2*)(sB32 + (kk & 1) * (NN * SB_STRIDE / 2) + n * (SB_STRIDE / 2) +
                  (cq << 1)) = v;
      }
#endif
      __syncthreads();

      // A fragment: lane l<16 -> row l, K = kb..kb+7 & kb+16..kb+23 (ISA layout)
      const unsigned short* ap =
          sA + ((m_tile << 4) + l16) * SA_STRIDE + (kk << 5) + (lhalf << 3);
      const v8u a0 = *(const v8u*)ap;
      const v8u a1 = *(const v8u*)(ap + 16);
      const bf16x16 Af = __builtin_bit_cast(bf16x16,
          __builtin_shufflevector(a0, a1, 0,1,2,3,4,5,6,7,8,9,10,11,12,13,14,15));

      #pragma unroll
      for (int t = 0; t < 6; ++t) {
        // B fragment: lane -> col l16, K = lhalf*8..+7 & lhalf*8+16..+23 (ISA layout)
        const unsigned short* bp =
            sBk + (((n_half * 6 + t) << 4) + l16) * SB_STRIDE + (lhalf << 3);
        const v8u bv0 = *(const v8u*)bp;
        const v8u bv1 = *(const v8u*)(bp + 16);
        const bf16x16 Bf = __builtin_bit_cast(bf16x16,
            __builtin_shufflevector(bv0, bv1, 0,1,2,3,4,5,6,7,8,9,10,11,12,13,14,15));
        cacc[t] = __builtin_amdgcn_wmma_f32_16x16x32_bf16(
            false, Af, false, Bf, (short)0, cacc[t], false, false);
      }
      __syncthreads();
    }

    // --- spill C to LDS (aliases A/B region; safe after barrier) ---
    #pragma unroll
    for (int t = 0; t < 6; ++t) {
      const int ncol  = ((n_half * 6 + t) << 4) + l16;
      const int mbase = (m_tile << 4) + (lhalf << 3);
      #pragma unroll
      for (int r = 0; r < 8; ++r)
        sScores[(mbase + r) * SS_STRIDE + ncol] = cacc[t][r];
    }
    __syncthreads();

    // --- epilogue: bias + LeakyReLU + Wh dot, var, NLL, masked-softmax mix ---
    const float xvv = xv[(size_t)(b0 + erow) * Dd + j];
    const float* wj = wsoft + j * (Kk * Kk);
    #pragma unroll
    for (int kq = 0; kq < 4; ++kq) {
      const int k = eq + (kq << 2);
      float macc = 0.0f, sacc = 0.0f;
      #pragma unroll
      for (int h = 0; h < Hh; ++h) {
        float hp = sScores[erow * SS_STRIDE + k * Hh + h] + b1[(j * Kk + k) * Hh + h];
        hp = (hp > 0.0f) ? hp : 0.2f * hp;
        macc += hp * Wh[(k * Dd + j) * Hh + h];
        float sp = sScores[erow * SS_STRIDE + 96 + k * Hh + h] + sb1[(j * Kk + k) * Hh + h];
        sp = (sp > 0.0f) ? sp : 0.2f * sp;
        sacc += sp * sWh[(k * Dd + j) * Hh + h];
      }
      const float mean = macc + sScores[erow * SS_STRIDE + 80 + k] + b2[j * Kk + k];
      const float shv  = sacc + sScores[erow * SS_STRIDE + 176 + k] + sb2[j * Kk + k];
      const float sig  = 10.0f / (1.0f + __expf(-shv));
      float var = sig * sig;
      var = (var < 1e-6f) ? 1e-6f : var;
      const float d = xvv - mean;
      const float nlln = 0.5f * (__logf(var) + d * d / var);
      for (int i = k; i < Kk; ++i)
        acc[i] += nlln * wj[i * Kk + k];
    }
    __syncthreads();   // scores/sA reusable next group
  }

  // --- cross-thread (4 per row) reduction of acc, then store nll(B,16) ---
  float* sPart = (float*)smem;                    // [4][64][16]
  #pragma unroll
  for (int i = 0; i < Kk; ++i)
    sPart[((eq << 6) + erow) * Kk + i] = acc[i];
  __syncthreads();
  const int o0 = tid << 2;
  #pragma unroll
  for (int o = o0; o < o0 + 4; ++o) {
    const int ro = o >> 4, ii = o & 15;
    float s = 0.0f;
    #pragma unroll
    for (int q = 0; q < 4; ++q) s += sPart[q * 1024 + ro * Kk + ii];
    out[(size_t)(b0 + ro) * Kk + ii] = s;
  }
}

// ---------------------------------------------------------------------------
extern "C" void kernel_launch(void* const* d_in, const int* in_sizes, int n_in,
                              void* d_out, int out_size, void* d_ws, size_t ws_size,
                              hipStream_t stream) {
  const float* x_values = (const float*)d_in[0];
  const float* x_inputs = (const float*)d_in[1];
  const float* targets  = (const float*)d_in[2];
  const float* W1  = (const float*)d_in[3];
  const float* b1  = (const float*)d_in[4];
  const float* Wh  = (const float*)d_in[5];
  const float* Wx  = (const float*)d_in[6];
  const float* b2  = (const float*)d_in[7];
  const float* sW1 = (const float*)d_in[8];
  const float* sb1 = (const float*)d_in[9];
  const float* sWh = (const float*)d_in[10];
  const float* sWx = (const float*)d_in[11];
  const float* sb2 = (const float*)d_in[12];

  // workspace: 6,291,456 B packed bf16 weights + 65,536 B softmax weights
  unsigned short* Wc    = (unsigned short*)d_ws;
  float*          wsoft = (float*)((char*)d_ws + (size_t)Dd * NN * INN * 2);

  prep_weights<<<dim3(Dd), dim3(256), 0, stream>>>(W1, Wx, sW1, sWx, Wc);
  prep_softmax<<<dim3(4),  dim3(256), 0, stream>>>(targets, wsoft);
  fused_mlp_nll<<<dim3(B_TOT / MT), dim3(256), 0, stream>>>(
      x_values, x_inputs, b1, Wh, b2, sb1, sWh, sb2, Wc, wsoft, (float*)d_out);
}
